// OwnRCNN_18141941859053
// MI455X (gfx1250) — compile-verified, matrix operand order checked by reading, not run on hardware
//
#include <hip/hip_runtime.h>
#include <cstdint>

// ---------------------------------------------------------------------------
// RCNN (bidirectional 2-layer LSTM + fusion + masked max-pool + FC) for
// MI455X / gfx1250.  GEMMs use v_wmma_f32_16x16x32_bf16 (wave32 WMMA);
// the sequential LSTM scan prefetches next-step gate biases into LDS with
// GLOBAL_LOAD_ASYNC_TO_LDS_B128 (ASYNCcnt) to hide L2 latency on the
// 512-step critical path.
// ---------------------------------------------------------------------------

typedef __attribute__((ext_vector_type(16))) __bf16 v16bf;
typedef __attribute__((ext_vector_type(8)))  float  v8f;

union BFrag { v16bf v; uint4 q[2]; unsigned short u[16]; };

static constexpr int Ee  = 256;        // embed dim
static constexpr int Hh  = 512;        // hidden
static constexpr int Bb  = 32;         // batch
static constexpr int Ss  = 512;        // seq len
static constexpr int Cc  = 5;          // classes
static constexpr int G4H = 4 * Hh;     // 2048 gate width
static constexpr int H2  = 2 * Hh;     // 1024
static constexpr int FK  = H2 + Ee;    // 1280 fusion K

#define HD __device__ __forceinline__

HD unsigned short f2bf(float f) {                       // RNE float->bf16
    unsigned u = __float_as_uint(f);
    unsigned r = u + 0x7fffu + ((u >> 16) & 1u);
    return (unsigned short)(r >> 16);
}
HD float sigm(float x) { return 1.0f / (1.0f + expf(-x)); }

// monotone order-preserving f32 <-> u32 encoding for atomicMax pooling
HD unsigned encf(float f) {
    unsigned u = __float_as_uint(f);
    return (u >> 31) ? ~u : (u | 0x80000000u);
}
HD float decf(unsigned e) {
    unsigned u = (e & 0x80000000u) ? (e & 0x7fffffffu) : ~e;
    return __uint_as_float(u);
}

// ---------------------------------------------------------------------------
// CDNA5 async global->LDS copy (tracked by ASYNCcnt).
// LDS destination offset = low 32 bits of the flat shared-memory address
// (ISA aperture mapping).
// ---------------------------------------------------------------------------
HD void async_copy_b128(const void* gptr, void* lds_ptr) {
    unsigned lds_off = (unsigned)(uintptr_t)lds_ptr;
    asm volatile("global_load_async_to_lds_b128 %0, %1, off"
                 :: "v"(lds_off), "v"(gptr)
                 : "memory");
}
HD void async_wait0() {
    asm volatile("s_wait_asynccnt 0x0" ::: "memory");
}

// ---------------------------------------------------------------------------
// small utility kernels
// ---------------------------------------------------------------------------
__global__ void k_f32_to_bf16(const float* __restrict__ src,
                              unsigned short* __restrict__ dst, int n) {
    int i = blockIdx.x * blockDim.x + threadIdx.x;
    if (i < n) dst[i] = f2bf(src[i]);
}

__global__ void k_embed_gather(const int* __restrict__ x,       // [B,S]
                               const float* __restrict__ emb,   // [V,E]
                               unsigned short* __restrict__ out) // [S,B,E] bf16
{
    int idx = blockIdx.x * blockDim.x + threadIdx.x;
    const int total = Ss * Bb * Ee;
    if (idx >= total) return;
    int e   = idx & (Ee - 1);
    int row = idx >> 8;            // Ee == 256
    int b   = row & (Bb - 1);
    int s   = row >> 5;            // Bb == 32
    int tok = x[b * Ss + s];
    out[idx] = f2bf(emb[(size_t)tok * Ee + e]);
}

__global__ void k_init_pooled(unsigned* __restrict__ p) {
    int i = blockIdx.x * blockDim.x + threadIdx.x;
    if (i < Bb * H2) p[i] = 0u;    // enc() of any real value is > 0
}

// ---------------------------------------------------------------------------
// Generic bf16 WMMA GEMM:  Y[M,N](f32) = A[M,K](bf16) * W[N,K](bf16)^T + bias
// block = 256 threads (8 waves); block tile 32(M) x 256(N); wave tile 16x64.
// K loop is register double-buffered so loads for k+1 overlap WMMAs for k.
// ---------------------------------------------------------------------------
HD void load_frag(BFrag& f, const unsigned short* base) {
    f.q[0] = *(const uint4*)base;
    f.q[1] = *(const uint4*)(base + 16);
}

__global__ __launch_bounds__(256) void k_gemm_bf16_f32(
    const unsigned short* __restrict__ A,
    const unsigned short* __restrict__ W,
    const float* __restrict__ bias,
    float* __restrict__ Y,
    int N, int K)
{
    const int lane  = threadIdx.x & 31;
    const int wid   = threadIdx.x >> 5;
    const int ln    = lane & 15;
    const int hi8   = (lane & 16) ? 8 : 0;   // M offset for C/D frags
    const int kh    = hi8;                   // K offset for A/B frags
    const int mbase = blockIdx.x * 32 + (wid & 1) * 16;
    const int nbase = blockIdx.y * 256 + (wid >> 1) * 64;

    const unsigned short* arow = A + (size_t)(mbase + ln) * K + kh;
    const unsigned short* wrow[4];
    for (int j = 0; j < 4; ++j)
        wrow[j] = W + (size_t)(nbase + j * 16 + ln) * K + kh;

    v8f acc[4];
    for (int j = 0; j < 4; ++j)
        for (int r = 0; r < 8; ++r) acc[j][r] = 0.0f;

    const int nk = K >> 5;
    BFrag a_c, w_c[4];
    load_frag(a_c, arow);
    for (int j = 0; j < 4; ++j) load_frag(w_c[j], wrow[j]);

    for (int kb = 0; kb < nk; ++kb) {
        const int kn = (kb + 1 < nk) ? (kb + 1) * 32 : 0;
        BFrag a_n, w_n[4];
        load_frag(a_n, arow + kn);
        for (int j = 0; j < 4; ++j) load_frag(w_n[j], wrow[j] + kn);
        for (int j = 0; j < 4; ++j)
            acc[j] = __builtin_amdgcn_wmma_f32_16x16x32_bf16(
                false, a_c.v, false, w_c[j].v, (short)0, acc[j], false, false);
        a_c = a_n;
        for (int j = 0; j < 4; ++j) w_c[j] = w_n[j];
    }

    for (int j = 0; j < 4; ++j) {
        int n    = nbase + j * 16 + ln;
        float bv = bias[n];
        for (int r = 0; r < 8; ++r) {
            int m = mbase + hi8 + r;
            Y[(size_t)m * N + n] = acc[j][r] + bv;
        }
    }
}

// ---------------------------------------------------------------------------
// LSTM scan. One block per direction (grid = 2), 1024 threads = 32 waves.
// Per step: gates[32,2048] = xg_staged[t] + h @ w_hh^T via WMMA bf16.
// Wave w owns h-columns 16w..16w+15 => owns all 4 gates for those columns,
// so c stays in registers; h round-trips through padded LDS.
// xg[t+1] (32x2048 f32 = 256KB) is async-staged into LDS during step t.
// Dynamic LDS: [ h bf16 32x520 | xg stage f32 32x2048 ] = 295,424 B (<320KB).
// ---------------------------------------------------------------------------
static constexpr int LDP       = 520;                        // padded h stride
static constexpr int H_BYTES   = 32 * LDP * 2;               // 33,280
static constexpr int XG_BYTES  = Bb * G4H * 4;               // 262,144
static constexpr int SCAN_SMEM = H_BYTES + XG_BYTES;         // 295,424

__global__ __launch_bounds__(1024) void k_lstm_scan(
    const float* __restrict__ xg_f,           // [S,B,4H]
    const float* __restrict__ xg_b,           // [S,B,4H]
    const unsigned short* __restrict__ whh_f, // [4H,H] bf16
    const unsigned short* __restrict__ whh_b, // [4H,H] bf16
    unsigned short* __restrict__ rnn)         // [S,B,2H] bf16 out
{
    extern __shared__ char smem[];
    unsigned short* h_lds = (unsigned short*)smem;
    float*          xg_s  = (float*)(smem + H_BYTES);

    const int dir = blockIdx.x;
    const float*          xg     = dir ? xg_b : xg_f;
    const unsigned short* whh    = dir ? whh_b : whh_f;
    const int             colofs = dir ? Hh : 0;

    const int tid  = threadIdx.x;
    const int lane = tid & 31;
    const int wid  = tid >> 5;                // 0..31 == h-column tile
    const int ln   = lane & 15;
    const int hi8  = (lane & 16) ? 8 : 0;
    const int kh   = hi8;

    for (int i = tid; i < 32 * LDP; i += blockDim.x) h_lds[i] = 0;

    float creg[2][8];
    for (int mt = 0; mt < 2; ++mt)
        for (int r = 0; r < 8; ++r) creg[mt][r] = 0.0f;

    // stage xg for the first timestep (16 x b128 per thread = 256KB total)
    {
        const int tt0 = dir ? (Ss - 1) : 0;
        const float* src = xg + (size_t)tt0 * Bb * G4H;
        for (int i = 0; i < 16; ++i) {
            int c = tid + i * 1024;           // 16-byte chunk index
            async_copy_b128(src + c * 4, xg_s + c * 4);
        }
    }

    for (int t = 0; t < Ss; ++t) {
        const int tt = dir ? (Ss - 1 - t) : t;

        async_wait0();        // this wave's staged xg chunks are in LDS
        __syncthreads();      // everyone's chunks + prev-step h are visible

        // gate accumulators initialized from staged xg[t]
        v8f acc[2][4];
        for (int mt = 0; mt < 2; ++mt)
            for (int q = 0; q < 4; ++q) {
                const int n = q * Hh + wid * 16 + ln;
                for (int r = 0; r < 8; ++r) {
                    int m = mt * 16 + hi8 + r;
                    acc[mt][q][r] = xg_s[m * G4H + n];
                }
            }

        __syncthreads();      // all waves done reading xg_s: safe to restage

        if (t + 1 < Ss) {     // overlap next step's xg fetch with the K loop
            const int tn = dir ? (Ss - 2 - t) : (t + 1);
            const float* src = xg + (size_t)tn * Bb * G4H;
            for (int i = 0; i < 16; ++i) {
                int c = tid + i * 1024;
                async_copy_b128(src + c * 4, xg_s + c * 4);
            }
        }

        // gates += h @ w_hh^T
        for (int kb = 0; kb < Hh / 32; ++kb) {
            BFrag a0, a1, w[4];
            {
                const unsigned short* p0 = h_lds + (size_t)ln * LDP + kb * 32 + kh;
                a0.q[0] = *(const uint4*)p0;
                a0.q[1] = *(const uint4*)(p0 + 16);
                const unsigned short* p1 =
                    h_lds + (size_t)(16 + ln) * LDP + kb * 32 + kh;
                a1.q[0] = *(const uint4*)p1;
                a1.q[1] = *(const uint4*)(p1 + 16);
            }
            for (int q = 0; q < 4; ++q) {
                const unsigned short* p =
                    whh + (size_t)(q * Hh + wid * 16 + ln) * Hh + kb * 32 + kh;
                w[q].q[0] = *(const uint4*)p;
                w[q].q[1] = *(const uint4*)(p + 16);
            }
            for (int q = 0; q < 4; ++q) {
                acc[0][q] = __builtin_amdgcn_wmma_f32_16x16x32_bf16(
                    false, a0.v, false, w[q].v, (short)0, acc[0][q], false, false);
                acc[1][q] = __builtin_amdgcn_wmma_f32_16x16x32_bf16(
                    false, a1.v, false, w[q].v, (short)0, acc[1][q], false, false);
            }
        }

        // cell / hidden update (wave-private c in registers)
        unsigned short hbits[2][8];
        for (int mt = 0; mt < 2; ++mt)
            for (int r = 0; r < 8; ++r) {
                float iv = sigm(acc[mt][0][r]);
                float fv = sigm(acc[mt][1][r]);
                float gv = tanhf(acc[mt][2][r]);
                float ov = sigm(acc[mt][3][r]);
                float cv = fv * creg[mt][r] + iv * gv;
                creg[mt][r] = cv;
                float hv = ov * tanhf(cv);
                unsigned short hb = f2bf(hv);
                hbits[mt][r] = hb;
                int m = mt * 16 + hi8 + r;
                rnn[((size_t)tt * Bb + m) * H2 + colofs + wid * 16 + ln] = hb;
            }

        __syncthreads();      // all waves finished reading h_lds this step
        for (int mt = 0; mt < 2; ++mt)
            for (int r = 0; r < 8; ++r) {
                int m = mt * 16 + hi8 + r;
                h_lds[(size_t)m * LDP + wid * 16 + ln] = hbits[mt][r];
            }
        // visibility of new h is provided by the barrier at the top of t+1
    }
}

// ---------------------------------------------------------------------------
// Fusion GEMM over virtual concat [h_f | emb | h_b] (K=1280), tanh epilogue,
// masked max-pool via encoded atomicMax into pooled[B,2H].  Double-buffered.
// ---------------------------------------------------------------------------
HD uint4 load_concat(const unsigned short* rnn1, const unsigned short* embq,
                     int row, int k8) {
    if (k8 < Hh)       return *(const uint4*)(rnn1 + (size_t)row * H2 + k8);
    if (k8 < Hh + Ee)  return *(const uint4*)(embq + (size_t)row * Ee + (k8 - Hh));
    return *(const uint4*)(rnn1 + (size_t)row * H2 + Hh + (k8 - Hh - Ee));
}

__global__ __launch_bounds__(256) void k_fusion_pool(
    const unsigned short* __restrict__ rnn1,  // [S*B, 2H] bf16
    const unsigned short* __restrict__ embq,  // [S*B, E]  bf16
    const unsigned short* __restrict__ W,     // [2H, FK]  bf16
    const float* __restrict__ bias,           // [2H]
    const int* __restrict__ mask,             // [B,S]
    unsigned* __restrict__ pooled)            // [B,2H] encoded
{
    const int lane  = threadIdx.x & 31;
    const int wid   = threadIdx.x >> 5;
    const int ln    = lane & 15;
    const int hi8   = (lane & 16) ? 8 : 0;
    const int kh    = hi8;
    const int mbase = blockIdx.x * 32 + (wid & 1) * 16;
    const int nbase = blockIdx.y * 256 + (wid >> 1) * 64;
    const int arow  = mbase + ln;

    const unsigned short* wrow[4];
    for (int j = 0; j < 4; ++j)
        wrow[j] = W + (size_t)(nbase + j * 16 + ln) * FK + kh;

    v8f acc[4];
    for (int j = 0; j < 4; ++j)
        for (int r = 0; r < 8; ++r) acc[j][r] = 0.0f;

    constexpr int nk = FK / 32;
    BFrag a_c, w_c[4];
    a_c.q[0] = load_concat(rnn1, embq, arow, kh);
    a_c.q[1] = load_concat(rnn1, embq, arow, kh + 16);
    for (int j = 0; j < 4; ++j) load_frag(w_c[j], wrow[j]);

    for (int kb = 0; kb < nk; ++kb) {
        const int kn = (kb + 1 < nk) ? (kb + 1) * 32 : 0;
        BFrag a_n, w_n[4];
        a_n.q[0] = load_concat(rnn1, embq, arow, kn + kh);
        a_n.q[1] = load_concat(rnn1, embq, arow, kn + kh + 16);
        for (int j = 0; j < 4; ++j) load_frag(w_n[j], wrow[j] + kn);
        for (int j = 0; j < 4; ++j)
            acc[j] = __builtin_amdgcn_wmma_f32_16x16x32_bf16(
                false, a_c.v, false, w_c[j].v, (short)0, acc[j], false, false);
        a_c = a_n;
        for (int j = 0; j < 4; ++j) w_c[j] = w_n[j];
    }

    for (int j = 0; j < 4; ++j) {
        int n    = nbase + j * 16 + ln;
        float bv = bias[n];
        for (int r = 0; r < 8; ++r) {
            int m = mbase + hi8 + r;
            int s = m >> 5;            // row = s*B + b, B==32
            int b = m & 31;
            float    v = tanhf(acc[j][r] + bv);
            unsigned e = mask[b * Ss + s] ? encf(v) : encf(-1e9f);
            atomicMax(&pooled[b * H2 + n], e);
        }
    }
}

// ---------------------------------------------------------------------------
// Final FC: out[B,C] = dec(pooled) @ fc_w^T + fc_b   (160 outputs)
// ---------------------------------------------------------------------------
__global__ void k_fc(const unsigned* __restrict__ pooled,
                     const float* __restrict__ fc_w,
                     const float* __restrict__ fc_b,
                     float* __restrict__ out) {
    int t = blockIdx.x * blockDim.x + threadIdx.x;
    if (t >= Bb * Cc) return;
    int b = t / Cc, c = t % Cc;
    float s = fc_b[c];
    for (int k = 0; k < H2; ++k)
        s += decf(pooled[b * H2 + k]) * fc_w[c * H2 + k];
    out[t] = s;
}

// ---------------------------------------------------------------------------
extern "C" void kernel_launch(void* const* d_in, const int* in_sizes, int n_in,
                              void* d_out, int out_size, void* d_ws, size_t ws_size,
                              hipStream_t stream)
{
    (void)in_sizes; (void)n_in; (void)out_size; (void)ws_size;

    const int*   x        = (const int*)  d_in[0];
    const int*   mask     = (const int*)  d_in[1];
    const float* emb      = (const float*)d_in[2];
    const float* fusion_w = (const float*)d_in[3];
    const float* fusion_b = (const float*)d_in[4];
    const float* fc_w     = (const float*)d_in[5];
    const float* fc_b     = (const float*)d_in[6];
    // order: f0, b0, f1, b1
    const float* w_ih[4] = { (const float*)d_in[7],  (const float*)d_in[10],
                             (const float*)d_in[13], (const float*)d_in[16] };
    const float* w_hh[4] = { (const float*)d_in[8],  (const float*)d_in[11],
                             (const float*)d_in[14], (const float*)d_in[17] };
    const float* b_g[4]  = { (const float*)d_in[9],  (const float*)d_in[12],
                             (const float*)d_in[15], (const float*)d_in[18] };

    char*  ws  = (char*)d_ws;
    size_t off = 0;
    auto take = [&](size_t bytes) -> char* {
        char* p = ws + off;
        off = (off + bytes + 255) & ~(size_t)255;
        return p;
    };

    const int wih_k[4] = { Ee, Ee, H2, H2 };
    unsigned short* wih_bf[4];
    for (int i = 0; i < 4; ++i) wih_bf[i] = (unsigned short*)take((size_t)G4H * wih_k[i] * 2);
    unsigned short* whh_bf[4];
    for (int i = 0; i < 4; ++i) whh_bf[i] = (unsigned short*)take((size_t)G4H * Hh * 2);
    unsigned short* fw_bf  = (unsigned short*)take((size_t)H2 * FK * 2);
    unsigned short* emb_bf = (unsigned short*)take((size_t)Ss * Bb * Ee * 2);
    unsigned short* rnn0   = (unsigned short*)take((size_t)Ss * Bb * H2 * 2);
    unsigned short* rnn1   = (unsigned short*)take((size_t)Ss * Bb * H2 * 2);
    float*          xg_f   = (float*)take((size_t)Ss * Bb * G4H * 4);
    float*          xg_b   = (float*)take((size_t)Ss * Bb * G4H * 4);
    unsigned*       pooled = (unsigned*)take((size_t)Bb * H2 * 4);

    auto conv = [&](const float* s, unsigned short* d, size_t n) {
        k_f32_to_bf16<<<(unsigned)((n + 255) / 256), 256, 0, stream>>>(s, d, (int)n);
    };
    for (int i = 0; i < 4; ++i) {
        conv(w_ih[i], wih_bf[i], (size_t)G4H * wih_k[i]);
        conv(w_hh[i], whh_bf[i], (size_t)G4H * Hh);
    }
    conv(fusion_w, fw_bf, (size_t)H2 * FK);

    k_embed_gather<<<(Ss * Bb * Ee + 255) / 256, 256, 0, stream>>>(x, emb, emb_bf);
    k_init_pooled<<<(Bb * H2 + 255) / 256, 256, 0, stream>>>(pooled);

    const dim3 gx(Ss * Bb / 32, G4H / 256);   // (512, 8)

    // layer 0: input projections (parallel over S*B), then scan
    k_gemm_bf16_f32<<<gx, 256, 0, stream>>>(emb_bf, wih_bf[0], b_g[0], xg_f, G4H, Ee);
    k_gemm_bf16_f32<<<gx, 256, 0, stream>>>(emb_bf, wih_bf[1], b_g[1], xg_b, G4H, Ee);
    k_lstm_scan<<<2, 1024, SCAN_SMEM, stream>>>(xg_f, xg_b, whh_bf[0], whh_bf[1], rnn0);

    // layer 1
    k_gemm_bf16_f32<<<gx, 256, 0, stream>>>(rnn0, wih_bf[2], b_g[2], xg_f, G4H, H2);
    k_gemm_bf16_f32<<<gx, 256, 0, stream>>>(rnn0, wih_bf[3], b_g[3], xg_b, G4H, H2);
    k_lstm_scan<<<2, 1024, SCAN_SMEM, stream>>>(xg_f, xg_b, whh_bf[2], whh_bf[3], rnn1);

    // fusion + tanh + masked max-pool
    const dim3 gf(Ss * Bb / 32, H2 / 256);    // (512, 4)
    k_fusion_pool<<<gf, 256, 0, stream>>>(rnn1, emb_bf, fw_bf, fusion_b, mask, pooled);

    // final classifier
    k_fc<<<1, 256, 0, stream>>>(pooled, fc_w, fc_b, (float*)d_out);
}
